// ResidualVQ_75720273428939
// MI455X (gfx1250) — matrix operand order repaired and checked by reading, not compiled, not run on hardware
//
#include <hip/hip_runtime.h>
#include <hip/hip_bf16.h>

// ---------------------------------------------------------------------------
// ResidualVQ forward for MI455X (gfx1250): bf16 WMMA + Tensor Data Mover.
// ---------------------------------------------------------------------------

typedef __attribute__((ext_vector_type(16))) __bf16 v16bf;
typedef __attribute__((ext_vector_type(8)))  float  v8f;
typedef __attribute__((ext_vector_type(4)))  unsigned int u32x4;
typedef __attribute__((ext_vector_type(4)))  int          i32x4;
typedef __attribute__((ext_vector_type(8)))  int          i32x8;

union FragAB { v16bf v; uint4 q[2]; };

constexpr int BATCH = 8;
constexpr int TLEN  = 2048;
constexpr int NTOK  = BATCH * TLEN;   // 16384 tokens
constexpr int CDIM  = 256;
constexpr int CSZ   = 1024;
constexpr int NQn   = 8;
constexpr int RVQn  = 512;
constexpr int DIN   = 768;

__device__ __forceinline__ unsigned short f32_to_bf16(float f) {
  unsigned int u = __float_as_uint(f);
  u += 0x7FFFu + ((u >> 16) & 1u);          // round-to-nearest-even
  return (unsigned short)(u >> 16);
}

__device__ __forceinline__ v8f wmma_bf16(const FragAB& a, const FragAB& b, v8f c) {
  return __builtin_amdgcn_wmma_f32_16x16x32_bf16(
      /*neg_a=*/false, a.v, /*neg_b=*/false, b.v,
      /*c_mod=*/(short)0, c, /*reuse_a=*/false, /*reuse_b=*/false);
}

// ---- Tensor Data Mover: 2-D bf16 tile (global -> LDS) with LDS row padding.
// D# layout per CDNA5 ISA 8.3-8.6. data_size = 2 bytes. type = 2 ("image").
// padInterval/padAmount codes per group1 bits [24:22]/[31:25].
// This toolchain exposes the 6-arg builtin (g0, g1, g2, g3, extra, cpol).
__device__ __forceinline__ void tdm_load_2d(unsigned ldsOff, const void* gsrc,
                                            unsigned tensorD0, unsigned tensorD1,
                                            unsigned tileD0,   unsigned tileD1,
                                            unsigned strideD0,
                                            unsigned padInterval, unsigned padAmount) {
  const unsigned long long ga = (unsigned long long)gsrc;
  u32x4 g0;
  g0[0] = 1u;                                          // count = 1 (valid D#)
  g0[1] = ldsOff;                                      // lds_addr (bytes)
  g0[2] = (unsigned)(ga & 0xFFFFFFFFu);                // global_addr[31:0]
  g0[3] = (unsigned)((ga >> 32) & 0x01FFFFFFu)         // global_addr[56:32]
        | (2u << 30);                                  // type = 2
  i32x8 g1;
  g1[0] = (int)((1u << 16)                             // data_size = 2 bytes
              | (1u << 20)                             // pad_enable
              | (padInterval << 22)
              | (padAmount << 25));
  g1[1] = (int)((tensorD0 & 0xFFFFu) << 16);           // tensor_dim0[15:0]
  g1[2] = (int)(((tensorD0 >> 16) & 0xFFFFu)           // tensor_dim0[31:16]
              | ((tensorD1 & 0xFFFFu) << 16));         // tensor_dim1[15:0]
  g1[3] = (int)(((tensorD1 >> 16) & 0xFFFFu)           // tensor_dim1[31:16]
              | ((tileD0 & 0xFFFFu) << 16));           // tile_dim0
  g1[4] = (int)(tileD1 & 0xFFFFu);                     // tile_dim1 (tile_dim2 = 0)
  g1[5] = (int)strideD0;                               // tensor_dim0_stride[31:0]
  g1[6] = 0;                                           // stride hi, dim1_stride lo
  g1[7] = 0;
  const i32x4 z4 = {0, 0, 0, 0};
  const i32x8 z8 = {0, 0, 0, 0, 0, 0, 0, 0};
  __builtin_amdgcn_tensor_load_to_lds(g0, g1, z4, z4, z8, 0);
}

// --------------------------- f32 -> bf16 convert ----------------------------
__global__ void cvt_bf16_k(const float* __restrict__ in,
                           unsigned short* __restrict__ out, int n) {
  int i = blockIdx.x * 256 + threadIdx.x;
  int stride = gridDim.x * 256;
  for (; i < n; i += stride) out[i] = f32_to_bf16(in[i]);
}

// --------------------------- codebook row norms -----------------------------
__global__ __launch_bounds__(128)
void cbnorm_k(const float* __restrict__ cb, float* __restrict__ norms, int nRows) {
  int row  = blockIdx.x * 4 + (threadIdx.x >> 5);
  int lane = threadIdx.x & 31;
  if (row >= nRows) return;
  const float* p = cb + (size_t)row * CDIM + lane * 8;
  float4 a = *(const float4*)p;
  float4 b = *(const float4*)(p + 4);
  float s = a.x*a.x + a.y*a.y + a.z*a.z + a.w*a.w
          + b.x*b.x + b.y*b.y + b.z*b.z + b.w*b.w;
  #pragma unroll
  for (int off = 16; off > 0; off >>= 1) s += __shfl_xor(s, off, 32);
  if (lane == 0) norms[row] = s;
}

// ------------------------------ tiled GEMM ----------------------------------
// Out[b, m, t] = sum_k W[m,k] * Bsrc[k, n] + bias[m],   n = b*TLEN + t
// Block: 128 threads (4 waves). Block tile: 64 (M) x 32 (N). K-step: 32.
// A tile (weights, bf16) arrives via TDM; B tile staged cooperatively
// (f32->bf16 convert, or bf16 codebook gather), overlapping the TDM transfer.
enum { M_XINIT = 0, M_ENC = 1, M_CONTRIB = 2, M_FINAL = 3 };

template <int MODE, int M, int K>
__global__ __launch_bounds__(128)
void gemm_k(const unsigned short* __restrict__ Wbf,   // M x K row-major bf16
            const float* __restrict__ bias,           // M
            const float* __restrict__ In0,            // (B, K, T) f32
            const float* __restrict__ In1,            // (B, K, T) f32 (FINAL)
            const unsigned short* __restrict__ cbbf,  // CS x CDIM bf16 (CONTRIB)
            const int* __restrict__ idx,              // NTOK (CONTRIB)
            float* __restrict__ Out0,                 // (B, M, T) f32
            float* __restrict__ Out1,                 // (B, M, T) f32 (XINIT)
            unsigned short* __restrict__ OutBf)       // NTOK x M bf16 (ENC)
{
  __shared__ __align__(16) unsigned short As[64 * 40];  // 64 rows x 32 K (+pad)
  __shared__ __align__(16) unsigned short Bs[32 * 40];  // 32 N   x 32 K (+pad)

  const int tid    = threadIdx.x;
  const int n0     = blockIdx.x * 32;     // token base; tiles never cross batch
  const int b      = n0 / TLEN;
  const int t0     = n0 % TLEN;
  const int mBlock = blockIdx.y * 64;
  const int lane   = tid & 31;
  const int waveId = tid >> 5;
  const int r      = lane & 15;
  const int g      = lane >> 4;           // lane half select

  const unsigned asOff = (unsigned)(size_t)(void*)As;   // LDS byte offset

  v8f acc0 = {0.f,0.f,0.f,0.f,0.f,0.f,0.f,0.f};
  v8f acc1 = {0.f,0.f,0.f,0.f,0.f,0.f,0.f,0.f};

  for (int kBase = 0; kBase < K; kBase += 32) {
    // ---- TDM: weight tile 64 x 32 bf16, LDS rows padded 32 -> 40 halves ----
    if (tid == 0)
      tdm_load_2d(asOff, Wbf + (size_t)mBlock * K + kBase,
                  /*tensorD0=*/32, /*tensorD1=*/64, /*tile0=*/32, /*tile1=*/64,
                  /*strideD0=*/(unsigned)K, /*padI(16 dw)=*/3, /*padA(4 dw)=*/3);

    // ---- stage B: 32 K x 32 N (stored N-major: Bs[n*40 + k]) ----
    if (MODE != M_CONTRIB) {
      const int k  = tid >> 2;            // 0..31
      const int nc = (tid & 3) * 8;       // 0,8,16,24
      const size_t goff = (size_t)b * K * TLEN + (size_t)(kBase + k) * TLEN + t0 + nc;
      float4 f0 = *(const float4*)(In0 + goff);
      float4 f1 = *(const float4*)(In0 + goff + 4);
      if (MODE == M_FINAL) {              // emb = x - residual, on the fly
        float4 r0 = *(const float4*)(In1 + goff);
        float4 r1 = *(const float4*)(In1 + goff + 4);
        f0.x -= r0.x; f0.y -= r0.y; f0.z -= r0.z; f0.w -= r0.w;
        f1.x -= r1.x; f1.y -= r1.y; f1.z -= r1.z; f1.w -= r1.w;
      }
      Bs[(nc + 0) * 40 + k] = f32_to_bf16(f0.x);
      Bs[(nc + 1) * 40 + k] = f32_to_bf16(f0.y);
      Bs[(nc + 2) * 40 + k] = f32_to_bf16(f0.z);
      Bs[(nc + 3) * 40 + k] = f32_to_bf16(f0.w);
      Bs[(nc + 4) * 40 + k] = f32_to_bf16(f1.x);
      Bs[(nc + 5) * 40 + k] = f32_to_bf16(f1.y);
      Bs[(nc + 6) * 40 + k] = f32_to_bf16(f1.z);
      Bs[(nc + 7) * 40 + k] = f32_to_bf16(f1.w);
    } else {                              // M_CONTRIB: gather codebook rows
      const int n  = tid >> 2;            // 0..31
      const int kc = (tid & 3) * 8;       // 0,8,16,24
      const int code = idx[n0 + n];
      *(uint4*)&Bs[n * 40 + kc] =
          *(const uint4*)(cbbf + (size_t)code * CDIM + kBase + kc);
    }
    if (tid == 0) __builtin_amdgcn_s_wait_tensorcnt(0);
    __syncthreads();

    // ---- fragments + WMMA (one A reused for two N sub-tiles) ----
    FragAB a, b0, b1;
    const unsigned short* arow = &As[(waveId * 16 + r) * 40];
    a.q[0] = *(const uint4*)&arow[8 * g];          // K = 8g .. 8g+7
    a.q[1] = *(const uint4*)&arow[16 + 8 * g];     // K = 16+8g .. 16+8g+7
    const unsigned short* br0 = &Bs[r * 40 + 8 * g];
    b0.q[0] = *(const uint4*)&br0[0];
    b0.q[1] = *(const uint4*)&br0[16];
    const unsigned short* br1 = &Bs[(16 + r) * 40 + 8 * g];
    b1.q[0] = *(const uint4*)&br1[0];
    b1.q[1] = *(const uint4*)&br1[16];
    acc0 = wmma_bf16(a, b0, acc0);
    acc1 = wmma_bf16(a, b1, acc1);
    __syncthreads();
  }

  // ---- epilogue ----
  const int mW = mBlock + waveId * 16 + (g ? 8 : 0);   // c[v] -> row mW+v
  #pragma unroll
  for (int j = 0; j < 2; j++) {
    const v8f cc = j ? acc1 : acc0;
    const int n  = n0 + j * 16 + r;
    const int t  = t0 + j * 16 + r;
    if (MODE == M_ENC) {
      uint4 pk;
      pk.x = f32_to_bf16(cc[0] + bias[mW + 0]) | ((unsigned)f32_to_bf16(cc[1] + bias[mW + 1]) << 16);
      pk.y = f32_to_bf16(cc[2] + bias[mW + 2]) | ((unsigned)f32_to_bf16(cc[3] + bias[mW + 3]) << 16);
      pk.z = f32_to_bf16(cc[4] + bias[mW + 4]) | ((unsigned)f32_to_bf16(cc[5] + bias[mW + 5]) << 16);
      pk.w = f32_to_bf16(cc[6] + bias[mW + 6]) | ((unsigned)f32_to_bf16(cc[7] + bias[mW + 7]) << 16);
      *(uint4*)&OutBf[(size_t)n * CDIM + mW] = pk;
    } else {
      #pragma unroll
      for (int v = 0; v < 8; v++) {
        const int m = mW + v;
        const size_t o = (size_t)b * M * TLEN + (size_t)m * TLEN + t;
        const float val = cc[v] + bias[m];
        if (MODE == M_XINIT)        { Out0[o] = val; Out1[o] = val; }
        else if (MODE == M_CONTRIB) { Out0[o] = Out0[o] - val; }   // residual -= contrib
        else                        { Out0[o] = val; }             // FINAL
      }
    }
  }
}

// --------------------- fused distance GEMM + argmin -------------------------
// dist(n,c) = ||cb_c||^2 - 2 * enc_n . cb_c   (enc-norm is row-constant)
// Block: 128 threads (4 waves) handles 64 tokens; loops all 1024 codes.
// Encoder panel + double-buffered codebook tiles arrive via TDM; the next
// tile's DMA overlaps the current tile's 8 WMMAs.
__global__ __launch_bounds__(128)
void dist_argmin_k(const unsigned short* __restrict__ enc,    // NTOK x CDIM bf16
                   const unsigned short* __restrict__ cbbf,   // CSZ  x CDIM bf16
                   const float* __restrict__ cbnorm,          // CSZ
                   int* __restrict__ idxOut,                  // NTOK
                   float* __restrict__ idxFloatOut)           // NTOK (into d_out)
{
  __shared__ __align__(16) unsigned short Ae[64 * 264];      // 64 tok x 256 (+pad)
  __shared__ __align__(16) unsigned short Bc[2][16 * 264];   // 2 x 16 codes x 256

  const int tid    = threadIdx.x;
  const int n0     = blockIdx.x * 64;
  const int lane   = tid & 31;
  const int waveId = tid >> 5;
  const int r      = lane & 15;
  const int g      = lane >> 4;

  const unsigned aeOff  = (unsigned)(size_t)(void*)Ae;
  const unsigned bcOff0 = (unsigned)(size_t)(void*)&Bc[0][0];
  const unsigned bcOff1 = (unsigned)(size_t)(void*)&Bc[1][0];

  if (tid == 0) {
    // enc panel: 64 rows x 256 halves, rows padded 256 -> 264 (128 dw + 4 dw)
    tdm_load_2d(aeOff, enc + (size_t)n0 * CDIM, 256, 64, 256, 64, 256,
                /*padI(128 dw)=*/6, /*padA(4 dw)=*/3);
    // first codebook tile: 16 rows x 256 halves
    tdm_load_2d(bcOff0, cbbf, 256, CSZ, 256, 16, 256, 6, 3);
    __builtin_amdgcn_s_wait_tensorcnt(1);   // in-order: enc panel complete
  }
  __syncthreads();

  // preload this wave's 8 A-fragments (K = 0..255 in steps of 32)
  FragAB afr[8];
  #pragma unroll
  for (int kk = 0; kk < 8; kk++) {
    const unsigned short* arow = &Ae[(waveId * 16 + r) * 264 + kk * 32];
    afr[kk].q[0] = *(const uint4*)&arow[8 * g];
    afr[kk].q[1] = *(const uint4*)&arow[16 + 8 * g];
  }

  float runv[8]; int runi[8];
  #pragma unroll
  for (int v = 0; v < 8; v++) { runv[v] = 3.0e38f; runi[v] = 0; }

  for (int ci = 0; ci < CSZ / 16; ci++) {
    const int cur = ci & 1;
    if (tid == 0) __builtin_amdgcn_s_wait_tensorcnt(0);   // Bc[cur] resident
    __syncthreads();
    if (tid == 0 && ci + 1 < CSZ / 16)                    // prefetch next tile
      tdm_load_2d(cur ? bcOff0 : bcOff1,
                  cbbf + (size_t)(ci + 1) * 16 * CDIM, 256, CSZ, 256, 16, 256, 6, 3);

    const unsigned short* bbase = &Bc[cur][0];
    v8f acc = {0.f,0.f,0.f,0.f,0.f,0.f,0.f,0.f};
    #pragma unroll
    for (int kk = 0; kk < 8; kk++) {
      FragAB bb;
      const unsigned short* brow = bbase + r * 264 + kk * 32 + 8 * g;
      bb.q[0] = *(const uint4*)&brow[0];
      bb.q[1] = *(const uint4*)&brow[16];
      acc = wmma_bf16(afr[kk], bb, acc);
    }

    const int code = ci * 16 + r;          // this lane's N column
    const float cn = cbnorm[code];
    #pragma unroll
    for (int v = 0; v < 8; v++) {
      const float d = cn - 2.0f * acc[v];
      if (d < runv[v]) { runv[v] = d; runi[v] = code; }
    }
  }

  // reduce across the 16 N-lanes of each half (ties -> lowest index)
  #pragma unroll
  for (int off = 1; off < 16; off <<= 1) {
    #pragma unroll
    for (int v = 0; v < 8; v++) {
      const float ov = __shfl_xor(runv[v], off, 32);
      const int   oi = __shfl_xor(runi[v], off, 32);
      if (ov < runv[v] || (ov == runv[v] && oi < runi[v])) { runv[v] = ov; runi[v] = oi; }
    }
  }
  if (r == 0) {
    #pragma unroll
    for (int v = 0; v < 8; v++) {
      const int token = n0 + waveId * 16 + v + (g ? 8 : 0);
      idxOut[token]      = runi[v];
      idxFloatOut[token] = (float)runi[v];
    }
  }
}

// ------------------------------- launch -------------------------------------
extern "C" void kernel_launch(void* const* d_in, const int* in_sizes, int n_in,
                              void* d_out, int out_size, void* d_ws, size_t ws_size,
                              hipStream_t stream) {
  (void)in_sizes; (void)n_in; (void)out_size;
  const float* z   = (const float*)d_in[0];
  const float* ipw = (const float*)d_in[1];
  const float* ipb = (const float*)d_in[2];
  const float* inw = (const float*)d_in[3];
  const float* inb = (const float*)d_in[4];
  const float* cb  = (const float*)d_in[5];
  const float* ow  = (const float*)d_in[6];
  const float* ob  = (const float*)d_in[7];
  const float* opw = (const float*)d_in[8];
  const float* opb = (const float*)d_in[9];
  float* out = (float*)d_out;

  // ---- carve workspace ----
  char* p = (char*)d_ws;
  auto carve = [&](size_t bytes) -> char* {
    char* r = p; p += (bytes + 255) & ~(size_t)255; return r;
  };
  float*          x     = (float*)carve((size_t)BATCH * RVQn * TLEN * 4);
  float*          resid = (float*)carve((size_t)BATCH * RVQn * TLEN * 4);
  unsigned short* encb  = (unsigned short*)carve((size_t)NTOK * CDIM * 2);
  unsigned short* cbbf  = (unsigned short*)carve((size_t)NQn * CSZ * CDIM * 2);
  float*          cbn   = (float*)carve((size_t)NQn * CSZ * 4);
  unsigned short* wI    = (unsigned short*)carve((size_t)RVQn * DIN * 2);
  unsigned short* wIn   = (unsigned short*)carve((size_t)NQn * CDIM * RVQn * 2);
  unsigned short* wO    = (unsigned short*)carve((size_t)NQn * RVQn * CDIM * 2);
  unsigned short* wOp   = (unsigned short*)carve((size_t)DIN * RVQn * 2);
  int*            idxb  = (int*)carve((size_t)NTOK * 4);
  if ((size_t)(p - (char*)d_ws) > ws_size) return;  // insufficient scratch

  // ---- per-call bf16 conversions + codebook norms ----
  auto cvt = [&](const float* src, unsigned short* dst, int n) {
    int grid = (n + 255) / 256; if (grid > 2048) grid = 2048;
    cvt_bf16_k<<<grid, 256, 0, stream>>>(src, dst, n);
  };
  cvt(ipw, wI,   RVQn * DIN);
  cvt(inw, wIn,  NQn * CDIM * RVQn);
  cvt(ow,  wO,   NQn * RVQn * CDIM);
  cvt(opw, wOp,  DIN * RVQn);
  cvt(cb,  cbbf, NQn * CSZ * CDIM);
  cbnorm_k<<<(NQn * CSZ) / 4, 128, 0, stream>>>(cb, cbn, NQn * CSZ);

  // ---- input_proj: x = W_in @ z + b ; residual = x ----
  gemm_k<M_XINIT, RVQn, DIN><<<dim3(NTOK / 32, RVQn / 64), 128, 0, stream>>>(
      wI, ipb, z, nullptr, nullptr, nullptr, x, resid, nullptr);

  // ---- 8 RVQ stages ----
  float* idxOutF = out + (size_t)BATCH * DIN * TLEN;   // indices tail of d_out
  for (int q = 0; q < NQn; q++) {
    gemm_k<M_ENC, CDIM, RVQn><<<dim3(NTOK / 32, CDIM / 64), 128, 0, stream>>>(
        wIn + (size_t)q * CDIM * RVQn, inb + q * CDIM, resid,
        nullptr, nullptr, nullptr, nullptr, nullptr, encb);
    dist_argmin_k<<<NTOK / 64, 128, 0, stream>>>(
        encb, cbbf + (size_t)q * CSZ * CDIM, cbn + q * CSZ,
        idxb, idxOutF + (size_t)q * NTOK);
    gemm_k<M_CONTRIB, RVQn, CDIM><<<dim3(NTOK / 32, RVQn / 64), 128, 0, stream>>>(
        wO + (size_t)q * RVQn * CDIM, ob + q * RVQn, nullptr, nullptr,
        cbbf + (size_t)q * CSZ * CDIM, idxb, resid, nullptr, nullptr);
  }

  // ---- output_proj on emb = x - residual ----
  gemm_k<M_FINAL, DIN, RVQn><<<dim3(NTOK / 32, DIN / 64), 128, 0, stream>>>(
      wOp, opb, x, resid, nullptr, nullptr, out, nullptr, nullptr);
}
